// decoder_7645041787040
// MI455X (gfx1250) — compile-verified, matrix operand order checked by reading, not run on hardware
//
#include <hip/hip_runtime.h>

// ---------------------------------------------------------------------------
// Types / WMMA helpers (CDNA5 gfx1250, wave32)
// ---------------------------------------------------------------------------
typedef _Float16 h16;
typedef __attribute__((ext_vector_type(8)))  _Float16 v8h;
typedef __attribute__((ext_vector_type(16))) _Float16 v16h;
typedef __attribute__((ext_vector_type(8)))  float    v8f;

static __device__ __forceinline__ v16h mk16(v8h lo, v8h hi) {
    return __builtin_shufflevector(lo, hi, 0,1,2,3,4,5,6,7,8,9,10,11,12,13,14,15);
}
static __device__ __forceinline__ v8f wmma32(v16h a, v16h b, v8f c) {
    // D = A(16x32,f16) * B(32x16,f16) + C(16x16,f32)
    return __builtin_amdgcn_wmma_f32_16x16x32_f16(false, a, false, b, (short)0, c, false, false);
}

// gfx1250 async global->LDS copy (ASYNCcnt-tracked), 16B per lane.
#define ASYNC_LDS 1
static __device__ __forceinline__ void async_copy16(h16* lds_ptr, const h16* gptr) {
#if ASYNC_LDS
    unsigned lds_off = (unsigned)(unsigned long)lds_ptr;   // low 32 bits = LDS byte offset
    asm volatile("global_load_async_to_lds_b128 %0, %1, off"
                 :: "v"(lds_off), "v"(gptr) : "memory");
#else
    *(v8h*)lds_ptr = *(const v8h*)gptr;
#endif
}
// Wait until at most `keep` async ops remain outstanding (0 = all done,
// 6 = the 6 just-issued next-tile copies may still be in flight).
static __device__ __forceinline__ void async_wait_keep(int keep) {
#if ASYNC_LDS
    if (keep == 6) asm volatile("s_wait_asynccnt 6" ::: "memory");
    else           asm volatile("s_wait_asynccnt 0" ::: "memory");
#else
    (void)keep;
#endif
}

// Model constants
#define BB    16
#define DD    768
#define HH    12
#define NBB   4
#define DEPTH 12
#define NN    785
#define SPB   196
#define LQ    197      // tokens per branch incl. cls
#define LP    224      // padded (14*16, 7*32)
#define FFD   3072
#define HDIM  64
#define MROWS (BB*NN) // 12560

// ---------------------------------------------------------------------------
// fp32 -> fp16 conversion (weights), grid-stride
// ---------------------------------------------------------------------------
__global__ void convert_f16_kernel(const float* __restrict__ src, h16* __restrict__ dst, long n) {
    long i = (long)blockIdx.x * blockDim.x + threadIdx.x;
    long stride = (long)gridDim.x * blockDim.x;
    for (; i < n; i += stride) dst[i] = (h16)src[i];
}

// x = x_in + pos_embed (broadcast over batch), writes fresh f32 buffer
__global__ void add_pos_kernel(const float* __restrict__ x, const float* __restrict__ pos,
                               float* __restrict__ out) {
    long total = (long)BB * NN * DD;
    long i = (long)blockIdx.x * blockDim.x + threadIdx.x;
    long stride = (long)gridDim.x * blockDim.x;
    for (; i < total; i += stride) out[i] = x[i] + pos[i % ((long)NN * DD)];
}

// ---------------------------------------------------------------------------
// LayerNorm: one 256-thread block per row of 768, writes f16
// ---------------------------------------------------------------------------
__global__ void __launch_bounds__(256)
ln_kernel(const float* __restrict__ x, const float* __restrict__ w, const float* __restrict__ bp,
          h16* __restrict__ out, float eps) {
    long row = blockIdx.x;
    const float* xr = x + row * DD;
    __shared__ float red[256];
    int t = threadIdx.x;
    float s = 0.f;
    for (int d = t; d < DD; d += 256) s += xr[d];
    red[t] = s; __syncthreads();
    for (int o = 128; o; o >>= 1) { if (t < o) red[t] += red[t + o]; __syncthreads(); }
    float mu = red[0] / (float)DD; __syncthreads();
    float v = 0.f;
    for (int d = t; d < DD; d += 256) { float df = xr[d] - mu; v += df * df; }
    red[t] = v; __syncthreads();
    for (int o = 128; o; o >>= 1) { if (t < o) red[t] += red[t + o]; __syncthreads(); }
    float rs = rsqrtf(red[0] / (float)DD + eps);
    for (int d = t; d < DD; d += 256)
        out[row * DD + d] = (h16)((xr[d] - mu) * rs * w[d] + bp[d]);
}

// Final LN over tokens 1..784, gathered into (g, b*196+s, d) layout for head GEMMs
__global__ void __launch_bounds__(256)
ln_gather_kernel(const float* __restrict__ x, const float* __restrict__ w,
                 const float* __restrict__ bp, h16* __restrict__ out) {
    int row = blockIdx.x;                 // 0..B*784-1
    int b = row / 784, tkn = row % 784;
    int g = tkn / SPB, s0 = tkn % SPB;
    const float* xr = x + ((long)b * NN + 1 + tkn) * DD;
    h16* orow = out + (((long)g * BB + b) * SPB + s0) * DD;
    __shared__ float red[256];
    int t = threadIdx.x;
    float s = 0.f;
    for (int d = t; d < DD; d += 256) s += xr[d];
    red[t] = s; __syncthreads();
    for (int o = 128; o; o >>= 1) { if (t < o) red[t] += red[t + o]; __syncthreads(); }
    float mu = red[0] / (float)DD; __syncthreads();
    float v = 0.f;
    for (int d = t; d < DD; d += 256) { float df = xr[d] - mu; v += df * df; }
    red[t] = v; __syncthreads();
    for (int o = 128; o; o >>= 1) { if (t < o) red[t] += red[t + o]; __syncthreads(); }
    float rs = rsqrtf(red[0] / (float)DD + 1e-5f);
    for (int d = t; d < DD; d += 256)
        orow[d] = (h16)((xr[d] - mu) * rs * w[d] + bp[d]);
}

// ---------------------------------------------------------------------------
// Generic WMMA GEMM: C(MxN) = A(MxK,f16) * Bt(NxK,f16)^T  [Bt row-major = W(out,in)]
// epi: 0 = store f16; 1 = +bias, f16; 2 = +bias +GELU(exact), f16;
//      3 = +bias +residual(f32 MxN), store f32; 4 = +bias, store f32
// Block: 128 threads / 4 waves, 64x128 tile, wave 32x64 (2x4 wmma), K step 32.
// Double-buffered LDS staged with global_load_async_to_lds_b128: tile k+1's
// copies (6 per wave) overlap tile k's WMMAs; s_wait_asynccnt 6 waits only
// for the previous tile (ASYNC ops retire in order).  N must be mult of 128.
// ---------------------------------------------------------------------------
__global__ void __launch_bounds__(128)
gemm_wmma_kernel(const h16* __restrict__ A, const h16* __restrict__ Bt,
                 const float* __restrict__ bias, const float* __restrict__ res,
                 void* __restrict__ outp, int M, int N, int K, int epi) {
    __shared__ h16 As[2][64 * 40];    // padded row stride 40 halves (80B, 16B-aligned)
    __shared__ h16 Bs[2][128 * 40];
    const int tid  = threadIdx.x;
    const int lane = tid & 31;
    const int wave = tid >> 5;
    const int wm = wave >> 1, wn = wave & 1;
    const int r  = lane & 15;
    const int hb = lane >> 4;
    const long brow = (long)blockIdx.x * 64;
    const long bcol = (long)blockIdx.y * 128;
    v8f acc[2][4];
    #pragma unroll
    for (int i = 0; i < 2; ++i)
        #pragma unroll
        for (int j = 0; j < 4; ++j) acc[i][j] = (v8f){};
    const int ksteps = K >> 5;

    auto load_tile = [&](int kt, int buf) {
        // A tile 64x32: 256 16B chunks, clamp OOB rows (masked at store)
        #pragma unroll
        for (int it = 0; it < 2; ++it) {
            int cid = tid + it * 128;
            int row = cid >> 2, cp = (cid & 3) * 8;
            long ga = brow + row; if (ga >= M) ga = M - 1;
            async_copy16(&As[buf][row * 40 + cp], A + ga * (long)K + kt * 32 + cp);
        }
        // B tile 128x32: 512 16B chunks (N multiple of 128 -> no guard)
        #pragma unroll
        for (int it = 0; it < 4; ++it) {
            int cid = tid + it * 128;
            int row = cid >> 2, cp = (cid & 3) * 8;
            async_copy16(&Bs[buf][row * 40 + cp], Bt + (bcol + row) * (long)K + kt * 32 + cp);
        }
    };

    load_tile(0, 0);
    for (int kt = 0; kt < ksteps; ++kt) {
        const int cur = kt & 1;
        if (kt + 1 < ksteps) {
            load_tile(kt + 1, cur ^ 1);   // overlap next copy with this compute
            async_wait_keep(6);           // previous tile's 6 copies done
        } else {
            async_wait_keep(0);
        }
        __syncthreads();                  // publish all waves' tile kt
        const h16* Ac = As[cur];
        const h16* Bc = Bs[cur];
        const int ra0 = (wm * 32 + r) * 40, ra1 = (wm * 32 + 16 + r) * 40;
        v16h a0 = mk16(*(const v8h*)(Ac + ra0 + hb * 8), *(const v8h*)(Ac + ra0 + 16 + hb * 8));
        v16h a1 = mk16(*(const v8h*)(Ac + ra1 + hb * 8), *(const v8h*)(Ac + ra1 + 16 + hb * 8));
        v16h bf[4];
        #pragma unroll
        for (int j = 0; j < 4; ++j) {
            const int rb = (wn * 64 + j * 16 + r) * 40;
            bf[j] = mk16(*(const v8h*)(Bc + rb + hb * 8), *(const v8h*)(Bc + rb + 16 + hb * 8));
        }
        #pragma unroll
        for (int j = 0; j < 4; ++j) {
            acc[0][j] = wmma32(a0, bf[j], acc[0][j]);
            acc[1][j] = wmma32(a1, bf[j], acc[1][j]);
        }
        __syncthreads();                  // reads done before buffer reuse
    }
    h16*   outh = (h16*)outp;
    float* outf = (float*)outp;
    #pragma unroll
    for (int i = 0; i < 2; ++i)
        #pragma unroll
        for (int j = 0; j < 4; ++j) {
            v8f cc = acc[i][j];
            long n = bcol + wn * 64 + j * 16 + r;
            float bv = (epi >= 1 && bias) ? bias[n] : 0.f;
            #pragma unroll
            for (int e = 0; e < 8; ++e) {
                long m = brow + wm * 32 + i * 16 + e + hb * 8;
                if (m < M) {
                    float v = cc[e] + bv;
                    if (epi == 2) v = 0.5f * v * (1.0f + erff(v * 0.70710678118f));
                    if (epi == 3) v += res[m * (long)N + n];
                    if (epi == 3 || epi == 4) outf[m * (long)N + n] = v;
                    else                      outh[m * (long)N + n] = (h16)v;
                }
            }
        }
}

// ---------------------------------------------------------------------------
// Global cls attention: one block per (b,h). q[:,:,0] over all 785 k/v.
// cls = attn + xn[:,0]; writes cls_h and attn row 0.
// ---------------------------------------------------------------------------
__global__ void __launch_bounds__(256)
cls_attn_kernel(const h16* __restrict__ qkv, const h16* __restrict__ xn,
                h16* __restrict__ cls, h16* __restrict__ attn) {
    int b = blockIdx.x / HH, h = blockIdx.x % HH;
    __shared__ float sc[NN];
    __shared__ float qv[HDIM];
    __shared__ float red[256];
    int t = threadIdx.x;
    if (t < HDIM) qv[t] = (float)qkv[((long)b * NN) * 2304 + h * HDIM + t];
    __syncthreads();
    for (int n = t; n < NN; n += 256) {
        const h16* kr = qkv + ((long)b * NN + n) * 2304 + DD + h * HDIM;
        float s = 0.f;
        for (int d = 0; d < HDIM; ++d) s += qv[d] * (float)kr[d];
        sc[n] = s * 0.125f;
    }
    __syncthreads();
    float lm = -3.4e38f;
    for (int n = t; n < NN; n += 256) lm = fmaxf(lm, sc[n]);
    red[t] = lm; __syncthreads();
    for (int o = 128; o; o >>= 1) { if (t < o) red[t] = fmaxf(red[t], red[t + o]); __syncthreads(); }
    float mx = red[0]; __syncthreads();
    float ls = 0.f;
    for (int n = t; n < NN; n += 256) { float e = expf(sc[n] - mx); sc[n] = e; ls += e; }
    red[t] = ls; __syncthreads();
    for (int o = 128; o; o >>= 1) { if (t < o) red[t] += red[t + o]; __syncthreads(); }
    float inv = 1.f / red[0];
    if (t < HDIM) {
        float acc = 0.f;
        const h16* vb = qkv + (long)b * NN * 2304 + 2 * DD + h * HDIM + t;
        for (int n = 0; n < NN; ++n) acc += sc[n] * (float)vb[(long)n * 2304];
        float val = acc * inv + (float)xn[((long)b * NN) * DD + h * HDIM + t];
        cls[(long)b * DD + h * HDIM + t] = (h16)val;
        attn[((long)b * NN) * DD + h * HDIM + t] = (h16)val;   // token 0 of attn output
    }
}

// Build V^T tiles: vt[bhg][d][t] (t padded to 224, t==0 is cls-v)
__global__ void build_vt_kernel(const h16* __restrict__ qkv, const h16* __restrict__ qkvc,
                                h16* __restrict__ vt) {
    int bhg = blockIdx.x;
    int g = bhg & 3, h = (bhg >> 2) % HH, b = bhg / (HH * NBB);
    long base = (long)bhg * HDIM * LP;
    for (int i = threadIdx.x; i < HDIM * LP; i += blockDim.x) {
        int d = i / LP, t = i % LP;
        h16 v = (h16)0.f;
        if (t == 0)        v = qkvc[(long)b * 2304 + 2 * DD + h * HDIM + d];
        else if (t < LQ)   v = qkv[((long)b * NN + 1 + g * SPB + (t - 1)) * 2304 + 2 * DD + h * HDIM + d];
        vt[base + i] = v;
    }
}

// Per-lane fragment gather for Q/K rows (token 0 = cls from qkv_c)
static __device__ __forceinline__ v16h
load_qk_frag(const h16* __restrict__ qkv, const h16* __restrict__ qkvc,
             int b, int g, int h, int tok, int off, int ks, int hb) {
    v8h lo = {}, hi = {};
    const h16* rowp = nullptr;
    if (tok == 0)      rowp = qkvc + (long)b * 2304 + off + h * HDIM;
    else if (tok < LQ) rowp = qkv + ((long)b * NN + 1 + g * SPB + (tok - 1)) * 2304 + off + h * HDIM;
    if (rowp) {
        lo = *(const v8h*)(rowp + ks * 32 + hb * 8);
        hi = *(const v8h*)(rowp + ks * 32 + 16 + hb * 8);
    }
    return mk16(lo, hi);
}

// S = Q*K^T * scale, per (b,h,g): 13x13 16x16 tiles, K=64 (2 wmma each)
__global__ void __launch_bounds__(128)
attn_scores_kernel(const h16* __restrict__ qkv, const h16* __restrict__ qkvc,
                   float* __restrict__ S) {
    int bhg = blockIdx.x;
    int g = bhg & 3, h = (bhg >> 2) % HH, b = bhg / (HH * NBB);
    int lane = threadIdx.x & 31, wave = threadIdx.x >> 5;
    int r = lane & 15, hb = lane >> 4;
    long base = (long)bhg * LP * LP;
    for (int tt = wave; tt < 13 * 13; tt += 4) {
        int mt = tt / 13, nt = tt % 13;
        v8f acc = {};
        #pragma unroll
        for (int ks = 0; ks < 2; ++ks) {
            v16h a  = load_qk_frag(qkv, qkvc, b, g, h, mt * 16 + r, 0,  ks, hb);
            v16h bf = load_qk_frag(qkv, qkvc, b, g, h, nt * 16 + r, DD, ks, hb);
            acc = wmma32(a, bf, acc);
        }
        #pragma unroll
        for (int e = 0; e < 8; ++e) {
            int m = mt * 16 + e + hb * 8;
            int n = nt * 16 + r;
            if (m < LQ) S[base + (long)m * LP + n] = acc[e] * 0.125f;
        }
    }
}

// Row softmax over first 197 cols; pads cols/rows with zeros (f16 probs, stride 224)
__global__ void __launch_bounds__(32)
softmax_kernel(const float* __restrict__ S, h16* __restrict__ P) {
    int rowid = blockIdx.x;            // bhg*224 + m
    int m = rowid % LP;
    long base = (long)(rowid / LP) * LP * LP + (long)m * LP;
    int t = threadIdx.x;
    if (m >= LQ) {
        for (int n = t; n < LP; n += 32) P[base + n] = (h16)0.f;
        return;
    }
    float mx = -3.4e38f;
    for (int n = t; n < LQ; n += 32) mx = fmaxf(mx, S[base + n]);
    for (int o = 16; o; o >>= 1) mx = fmaxf(mx, __shfl_xor(mx, o));
    float sum = 0.f;
    for (int n = t; n < LQ; n += 32) sum += expf(S[base + n] - mx);
    for (int o = 16; o; o >>= 1) sum += __shfl_xor(sum, o);
    float inv = 1.f / sum;
    for (int n = t; n < LP; n += 32) {
        float v = (n < LQ) ? expf(S[base + n] - mx) * inv : 0.f;
        P[base + n] = (h16)v;
    }
}

// O = P * V (via V^T), drop cls row, scatter with the reference's per-branch
// (H, spb, HD)->(spb, C) interleave.
__global__ void __launch_bounds__(128)
attn_av_kernel(const h16* __restrict__ P, const h16* __restrict__ Vt, h16* __restrict__ attn) {
    int bhg = blockIdx.x;
    int g = bhg & 3, h = (bhg >> 2) % HH, b = bhg / (HH * NBB);
    int lane = threadIdx.x & 31, wave = threadIdx.x >> 5;
    int r = lane & 15, hb = lane >> 4;
    long pbase = (long)bhg * LP * LP;
    long vbase = (long)bhg * HDIM * LP;
    for (int tt = wave; tt < 13 * 4; tt += 4) {   // 13 row tiles x 4 col tiles
        int mt = tt >> 2, nt = tt & 3;
        v8f acc = {};
        #pragma unroll
        for (int ks = 0; ks < 7; ++ks) {          // K = 224
            long ap = pbase + (long)(mt * 16 + r) * LP + ks * 32;
            v16h a  = mk16(*(const v8h*)(P + ap + hb * 8), *(const v8h*)(P + ap + 16 + hb * 8));
            long bp = vbase + (long)(nt * 16 + r) * LP + ks * 32;
            v16h bf = mk16(*(const v8h*)(Vt + bp + hb * 8), *(const v8h*)(Vt + bp + 16 + hb * 8));
            acc = wmma32(a, bf, acc);
        }
        #pragma unroll
        for (int e = 0; e < 8; ++e) {
            int m = mt * 16 + e + hb * 8;
            if (m >= 1 && m < LQ) {
                int s = m - 1, d = nt * 16 + r;
                long idx = (long)b * NN * DD + DD + (long)g * SPB * DD
                         + (long)h * SPB * HDIM + (long)s * HDIM + d;
                attn[idx] = (h16)acc[e];
            }
        }
    }
}

// Patch head output -> image layout (b, NB*3, 224, 224)
__global__ void head_rearrange_kernel(const float* __restrict__ y, float* __restrict__ out) {
    long idx = (long)blockIdx.x * blockDim.x + threadIdx.x;
    if (idx >= (long)BB * 12 * 224 * 224) return;
    int col = (int)(idx % 224); long t = idx / 224;
    int rowp = (int)(t % 224); t /= 224;
    int gc = (int)(t % 12); int b = (int)(t / 12);
    int g = gc / 3, c = gc % 3;
    int hy = rowp >> 4, p = rowp & 15;
    int wx = col >> 4, qq = col & 15;
    int s = hy * 14 + wx;
    int o = (p * 16 + qq) * 3 + c;
    out[idx] = y[(((long)g * BB + b) * SPB + s) * DD + o];
}

// ---------------------------------------------------------------------------
// Host orchestration
// ---------------------------------------------------------------------------
extern "C" void kernel_launch(void* const* d_in, const int* in_sizes, int n_in,
                              void* d_out, int out_size, void* d_ws, size_t ws_size,
                              hipStream_t stream) {
    (void)in_sizes; (void)n_in; (void)out_size; (void)ws_size;
    const float* in_x   = (const float*)d_in[0];
    const float* pos    = (const float*)d_in[1];
    const float* ln1_w  = (const float*)d_in[2];
    const float* ln1_b  = (const float*)d_in[3];
    const float* qkv_w  = (const float*)d_in[4];
    const float* proj_w = (const float*)d_in[5];
    const float* proj_b = (const float*)d_in[6];
    const float* ln2_w  = (const float*)d_in[7];
    const float* ln2_b  = (const float*)d_in[8];
    const float* fc1_w  = (const float*)d_in[9];
    const float* fc1_b  = (const float*)d_in[10];
    const float* fc2_w  = (const float*)d_in[11];
    const float* fc2_b  = (const float*)d_in[12];
    const float* dn_w   = (const float*)d_in[13];
    const float* dn_b   = (const float*)d_in[14];
    const float* head_w = (const float*)d_in[15];
    const float* head_b = (const float*)d_in[16];

    char* ws = (char*)d_ws;
    size_t off = 0;
    auto alloc = [&](size_t elems, size_t esz) -> void* {
        void* p = ws + off;
        off += ((elems * esz + 255) / 256) * 256;
        return p;
    };
    float* x_f    = (float*)alloc((size_t)BB * NN * DD, 4);
    h16*   xn_h   = (h16*)  alloc((size_t)BB * NN * DD, 2);
    h16*   qkv_h  = (h16*)  alloc((size_t)BB * NN * 3 * DD, 2);
    h16*   qkvc_h = (h16*)  alloc((size_t)BB * 3 * DD, 2);
    h16*   cls_h  = (h16*)  alloc((size_t)BB * DD, 2);
    h16*   vt_h   = (h16*)  alloc((size_t)BB * HH * NBB * HDIM * LP, 2);
    float* S_f    = (float*)alloc((size_t)BB * HH * NBB * LP * LP, 4);
    h16*   P_h    = (h16*)  alloc((size_t)BB * HH * NBB * LP * LP, 2);
    h16*   attn_h = (h16*)  alloc((size_t)BB * NN * DD, 2);
    h16*   hid_h  = (h16*)  alloc((size_t)BB * NN * FFD, 2);
    h16*   xg_h   = (h16*)  alloc((size_t)NBB * BB * SPB * DD, 2);
    float* y_f    = (float*)alloc((size_t)NBB * BB * SPB * DD, 4);
    h16*   qkvw_h = (h16*)  alloc((size_t)DEPTH * 3 * DD * DD, 2);
    h16*   projw_h= (h16*)  alloc((size_t)DEPTH * DD * DD, 2);
    h16*   fc1w_h = (h16*)  alloc((size_t)DEPTH * FFD * DD, 2);
    h16*   fc2w_h = (h16*)  alloc((size_t)DEPTH * DD * FFD, 2);
    h16*   headw_h= (h16*)  alloc((size_t)NBB * DD * DD, 2);

    // Weight conversions (deterministic, every call)
    convert_f16_kernel<<<4096, 256, 0, stream>>>(qkv_w,  qkvw_h, (long)DEPTH * 3 * DD * DD);
    convert_f16_kernel<<<4096, 256, 0, stream>>>(proj_w, projw_h,(long)DEPTH * DD * DD);
    convert_f16_kernel<<<4096, 256, 0, stream>>>(fc1_w,  fc1w_h, (long)DEPTH * FFD * DD);
    convert_f16_kernel<<<4096, 256, 0, stream>>>(fc2_w,  fc2w_h, (long)DEPTH * DD * FFD);
    convert_f16_kernel<<<4096, 256, 0, stream>>>(head_w, headw_h,(long)NBB * DD * DD);

    add_pos_kernel<<<4096, 256, 0, stream>>>(in_x, pos, x_f);

    auto gemm = [&](const h16* A, const h16* Bt, const float* bias, const float* res,
                    void* out, int M, int N, int K, int epi) {
        dim3 gsz((M + 63) / 64, N / 128);
        gemm_wmma_kernel<<<gsz, 128, 0, stream>>>(A, Bt, bias, res, out, M, N, K, epi);
    };

    const int BHG = BB * HH * NBB;   // 768
    for (int l = 0; l < DEPTH; ++l) {
        ln_kernel<<<MROWS, 256, 0, stream>>>(x_f, ln1_w + l * DD, ln1_b + l * DD, xn_h, 1e-6f);
        gemm(xn_h, qkvw_h + (size_t)l * 3 * DD * DD, nullptr, nullptr, qkv_h, MROWS, 3 * DD, DD, 0);
        cls_attn_kernel<<<BB * HH, 256, 0, stream>>>(qkv_h, xn_h, cls_h, attn_h);
        gemm(cls_h, qkvw_h + (size_t)l * 3 * DD * DD, nullptr, nullptr, qkvc_h, BB, 3 * DD, DD, 0);
        build_vt_kernel<<<BHG, 256, 0, stream>>>(qkv_h, qkvc_h, vt_h);
        attn_scores_kernel<<<BHG, 128, 0, stream>>>(qkv_h, qkvc_h, S_f);
        softmax_kernel<<<BHG * LP, 32, 0, stream>>>(S_f, P_h);
        attn_av_kernel<<<BHG, 128, 0, stream>>>(P_h, vt_h, attn_h);
        gemm(attn_h, projw_h + (size_t)l * DD * DD, proj_b + l * DD, x_f, x_f, MROWS, DD, DD, 3);
        ln_kernel<<<MROWS, 256, 0, stream>>>(x_f, ln2_w + l * DD, ln2_b + l * DD, xn_h, 1e-6f);
        gemm(xn_h, fc1w_h + (size_t)l * FFD * DD, fc1_b + l * FFD, nullptr, hid_h, MROWS, FFD, DD, 2);
        gemm(hid_h, fc2w_h + (size_t)l * DD * FFD, fc2_b + l * DD, x_f, x_f, MROWS, DD, FFD, 3);
    }

    ln_gather_kernel<<<BB * 784, 256, 0, stream>>>(x_f, dn_w, dn_b, xg_h);
    for (int g = 0; g < NBB; ++g)
        gemm(xg_h + (size_t)g * BB * SPB * DD, headw_h + (size_t)g * DD * DD,
             head_b + g * DD, nullptr, y_f + (size_t)g * BB * SPB * DD,
             BB * SPB, DD, DD, 4);
    long total = (long)BB * 12 * 224 * 224;
    head_rearrange_kernel<<<(unsigned)((total + 255) / 256), 256, 0, stream>>>(y_f, (float*)d_out);
}
